// Attention_42949672960606
// MI455X (gfx1250) — matrix operand order repaired and checked by reading, not compiled, not run on hardware
//
#include <hip/hip_runtime.h>
#include <hip/hip_bf16.h>

// ---------------------------------------------------------------------------
// Gated attention (AlphaFold-style) for MI455X / gfx1250, bf16 WMMA pipeline.
// Streaming tensors (bias, q/kv activations, final out) use non-temporal
// cache hints so the 268MB bias stream does not evict reused K/V/W tiles
// from the 192MB L2.
// ---------------------------------------------------------------------------

typedef __bf16 bf16_t;
typedef __attribute__((ext_vector_type(16))) __bf16 v16bf;
typedef __attribute__((ext_vector_type(8)))  float  v8f;
typedef __attribute__((ext_vector_type(4)))  float  v4f;   // clang vector: OK for NT builtins

#define NHEAD  8
#define CDIM   32
#define SEQ    256
#define NB2    128
#define DMODEL 256
#define MTOT   (NB2 * SEQ)      // 32768 rows
#define PSTR   264              // padded LDS row stride (bf16 elems): 528B = 33*16B

union FragU { v16bf v; uint4 q[2]; };

// A-fragment (16x32 bf16): per ISA, lane l elem e -> row l%16,
// K = (e%8) + 8*(l/16) + 16*(e/8). Caller passes base = row*stride + 8*(l/16);
// chunk0 at +0 (16B), chunk1 at +16 elems (+32B).
__device__ __forceinline__ v16bf afrag_bf16(const bf16_t* p16) {
  FragU f;
  const uint4* p = (const uint4*)p16;
  f.q[0] = p[0];
  f.q[1] = p[2];
  return f.v;
}

// B-fragment (32x16 bf16): lane l elem e -> col l%16,
// K = (e%8) + 8*(l/16) + 16*(e/8). Caller passes base = col_row*stride + 8*(l/16);
// chunk0 at +0 (16B), chunk1 at +16 elems (+32B).
__device__ __forceinline__ v16bf bfrag_bf16(const bf16_t* p16) {
  FragU f;
  const uint4* p = (const uint4*)p16;
  f.q[0] = p[0];
  f.q[1] = p[2];
  return f.v;
}

// A-fragment sourced from f32 memory (single-use stream): non-temporal
// 16B vector loads + on-the-fly bf16 convert.
__device__ __forceinline__ v16bf afrag_f32_nt(const float* p) {
  const v4f* q = (const v4f*)p;
  const v4f a0 = __builtin_nontemporal_load(q + 0);
  const v4f a1 = __builtin_nontemporal_load(q + 1);
  const v4f b0 = __builtin_nontemporal_load(q + 4);  // +16 floats
  const v4f b1 = __builtin_nontemporal_load(q + 5);
  v16bf r;
#pragma unroll
  for (int i = 0; i < 4; ++i) {
    r[i]      = (bf16_t)a0[i];
    r[4 + i]  = (bf16_t)a1[i];
    r[8 + i]  = (bf16_t)b0[i];
    r[12 + i] = (bf16_t)b1[i];
  }
  return r;
}

__device__ __forceinline__ v8f wmma_bf16(v16bf a, v16bf b, v8f c) {
  // D = A * B + C, f32 accumulate. (neg_a, A, neg_b, B, c_mod, C, reuse_a, reuse_b)
  return __builtin_amdgcn_wmma_f32_16x16x32_bf16(false, a, false, b, (short)0, c,
                                                 false, false);
}

// ---------------------------------------------------------------------------
// Kernel 0: convert weights to bf16; fold q-scale (c^-1/2) into Wq.
// Layout in dst: [Wq | Wk | Wv | Wg | Wo], each 256x256.
// ---------------------------------------------------------------------------
__global__ __launch_bounds__(256) void k_cvt(
    const float* __restrict__ wq, const float* __restrict__ wk,
    const float* __restrict__ wv, const float* __restrict__ wg,
    const float* __restrict__ wo, bf16_t* __restrict__ dst) {
  int i = blockIdx.x * 256 + threadIdx.x;
  if (i >= 5 * 65536) return;
  int m = i >> 16, r = i & 65535;
  const float* src = (m == 0) ? wq : (m == 1) ? wk : (m == 2) ? wv
                     : (m == 3) ? wg : wo;
  float s = (m == 0) ? 0.17677669529663687f : 1.0f;  // 1/sqrt(32)
  dst[i] = (bf16_t)(src[r] * s);
}

// ---------------------------------------------------------------------------
// Kernel 1: fused Q/K/V/Gate projections.
// Each wave owns 16 rows of the [32768, 256] activations and produces all
// 4 * 256 output columns with v_wmma_f32_16x16x32_bf16. Outputs stored as
// [b2][h][len][c] (bf16 for q/k/v, f32 sigmoid gate).
// ---------------------------------------------------------------------------
template <int MAT>
__device__ __forceinline__ void proj_one(
    const v16bf (&afr)[8], const bf16_t* __restrict__ W,
    const float* __restrict__ bg, bf16_t* __restrict__ o_bf,
    float* __restrict__ o_f32, int b2, int lq, int l16, int hl) {
#pragma unroll 1
  for (int nt = 0; nt < 16; ++nt) {
    v8f acc = {};
    const bf16_t* wb = W + (nt * 16 + l16) * DMODEL + 8 * hl;
#pragma unroll
    for (int kc = 0; kc < 8; ++kc)
      acc = wmma_bf16(afr[kc], bfrag_bf16(wb + kc * 32), acc);
    const int n = nt * 16 + l16;
    const int h = n >> 5, c = n & 31;
    const float bgv = (MAT == 3) ? bg[n] : 0.0f;
#pragma unroll
    for (int j = 0; j < 8; ++j) {
      const int r = lq + j + 8 * hl;  // D-layout: row = j + 8*(lane/16)
      const size_t off = ((size_t)(b2 * NHEAD + h) * SEQ + r) * CDIM + c;
      const float x = acc[j];
      if (MAT == 3) o_f32[off] = 1.0f / (1.0f + __expf(-(x + bgv)));
      else          o_bf[off]  = (bf16_t)x;
    }
  }
}

__global__ __launch_bounds__(128) void k_proj(
    const float* __restrict__ qd, const float* __restrict__ kvd,
    const bf16_t* __restrict__ wq, const bf16_t* __restrict__ wk,
    const bf16_t* __restrict__ wv, const bf16_t* __restrict__ wg,
    const float* __restrict__ bg,
    bf16_t* __restrict__ qo, bf16_t* __restrict__ ko,
    bf16_t* __restrict__ vo, float* __restrict__ go) {
  const int tid = threadIdx.x, lane = tid & 31, w = tid >> 5;
  const int l16 = lane & 15, hl = lane >> 4;
  const int m0 = blockIdx.x * 64 + w * 16;   // 16-row tile base
  const int row = m0 + l16;
  const int b2 = m0 >> 8, lq = m0 & 255;

  v16bf aq[8], akv[8];
#pragma unroll
  for (int kc = 0; kc < 8; ++kc) {
    aq[kc]  = afrag_f32_nt(qd  + (size_t)row * DMODEL + kc * 32 + 8 * hl);
    akv[kc] = afrag_f32_nt(kvd + (size_t)row * DMODEL + kc * 32 + 8 * hl);
  }
  proj_one<0>(aq,  wq, bg, qo, nullptr, b2, lq, l16, hl);  // q (pre-scaled)
  proj_one<1>(akv, wk, bg, ko, nullptr, b2, lq, l16, hl);  // k
  proj_one<2>(akv, wv, bg, vo, nullptr, b2, lq, l16, hl);  // v
  proj_one<3>(aq,  wg, bg, nullptr, go, b2, lq, l16, hl);  // sigmoid gate
}

// ---------------------------------------------------------------------------
// Kernel 2: attention core. Block = 4 waves, grid (4 lq-blocks, 8 heads, 128).
// Wave: 16 q-rows. 16 WMMAs -> full 16x256 logits in registers; bias
// streamed once with NT loads (+ prefetch); softmax via shfl_xor over the
// 16-lane halves; P staged to padded LDS; V staged transposed to LDS once per
// block; 16 WMMAs for P.V; gate multiply; bf16 store of weighted avg.
// ---------------------------------------------------------------------------
__global__ __launch_bounds__(128) void k_attn(
    const bf16_t* __restrict__ qm, const bf16_t* __restrict__ km,
    const bf16_t* __restrict__ vm, const float* __restrict__ gate,
    const float* __restrict__ bias, const float* __restrict__ nbias,
    bf16_t* __restrict__ wavg) {
  __shared__ bf16_t ldsP[4][16 * PSTR];   // per-wave probability tile
  __shared__ bf16_t ldsVT[CDIM * PSTR];   // V transposed: [c][lk], padded

  const int tid = threadIdx.x, lane = tid & 31, w = tid >> 5;
  const int l16 = lane & 15, hl = lane >> 4;
  const int b2 = blockIdx.z, h = blockIdx.y;
  const int lq0 = blockIdx.x * 64 + w * 16;
  const size_t bh = (size_t)(b2 * NHEAD + h);

  // Stage V[256][32] -> LDS transposed [32][256] (coalesced global reads).
  const bf16_t* vsrc = vm + bh * (SEQ * CDIM);
  for (int idx = tid; idx < SEQ * CDIM; idx += 128) {
    const int lk = idx >> 5, c = idx & 31;
    ldsVT[c * PSTR + lk] = vsrc[idx];
  }
  __syncthreads();

  // Q A-fragment: K-dim = C = 32 -> single fragment.
  const v16bf qf = afrag_bf16(qm + (bh * SEQ + lq0 + l16) * CDIM + 8 * hl);

  const float* brow = bias  + (bh * SEQ + lq0) * (size_t)SEQ;
  const float* nrow = nbias + ((size_t)h * SEQ + lq0) * SEQ;

  // Prefetch this wave's 16KB bias slab (gfx1250 global_prefetch_b8 path).
#pragma unroll
  for (int pr = 0; pr < 4; ++pr)
    __builtin_prefetch(brow + pr * 1024 + lane * 32, 0, 0);

  // Logits: 16 n-tiles of 16 columns, one WMMA each (K = 32).
  v8f acc[16];
#pragma unroll
  for (int nt = 0; nt < 16; ++nt) {
    const v16bf kf = bfrag_bf16(km + (bh * SEQ + nt * 16 + l16) * CDIM + 8 * hl);
    v8f a = {};
    a = wmma_bf16(qf, kf, a);
#pragma unroll
    for (int j = 0; j < 8; ++j) {
      const int r = j + 8 * hl, n = nt * 16 + l16;
      // bias: 268MB single-use stream -> NT; nbias: 2MB reused 128x -> RT.
      a[j] += __builtin_nontemporal_load(&brow[(size_t)r * SEQ + n]) +
              nrow[(size_t)r * SEQ + n];
    }
    acc[nt] = a;
  }

  // Softmax over the 256 columns of each row. Row m = j + 8*(lane/16) is
  // spread over the 16 lanes of this lane-half across the 16 tiles.
#pragma unroll
  for (int j = 0; j < 8; ++j) {
    float m = -3.0e38f;
#pragma unroll
    for (int nt = 0; nt < 16; ++nt) m = fmaxf(m, acc[nt][j]);
#pragma unroll
    for (int s = 1; s < 16; s <<= 1) m = fmaxf(m, __shfl_xor(m, s, 32));
    float sum = 0.0f;
#pragma unroll
    for (int nt = 0; nt < 16; ++nt) {
      const float e = __expf(acc[nt][j] - m);
      acc[nt][j] = e;
      sum += e;
    }
#pragma unroll
    for (int s = 1; s < 16; s <<= 1) sum += __shfl_xor(sum, s, 32);
    const float inv = 1.0f / sum;
#pragma unroll
    for (int nt = 0; nt < 16; ++nt) acc[nt][j] *= inv;
  }

  // P -> LDS (bf16), then P[16x256] x V[256x32] via 8 K-chunks x 2 n-tiles.
  bf16_t* P = &ldsP[w][0];
#pragma unroll
  for (int nt = 0; nt < 16; ++nt)
#pragma unroll
    for (int j = 0; j < 8; ++j)
      P[(j + 8 * hl) * PSTR + nt * 16 + l16] = (bf16_t)acc[nt][j];

  v8f o0 = {}, o1 = {};
#pragma unroll 1
  for (int kc = 0; kc < 8; ++kc) {
    const v16bf af = afrag_bf16(P + l16 * PSTR + kc * 32 + 8 * hl);
    const v16bf b0 = bfrag_bf16(&ldsVT[l16 * PSTR + kc * 32 + 8 * hl]);
    const v16bf b1 = bfrag_bf16(&ldsVT[(16 + l16) * PSTR + kc * 32 + 8 * hl]);
    o0 = wmma_bf16(af, b0, o0);
    o1 = wmma_bf16(af, b1, o1);
  }

  // Gate multiply + store weighted average as bf16 [b2*256+lq][h*32+c].
  const float* grow = gate + (bh * SEQ + lq0) * CDIM;
  bf16_t* orow = wavg + ((size_t)(b2 * SEQ + lq0)) * DMODEL + h * CDIM;
#pragma unroll
  for (int j = 0; j < 8; ++j) {
    const int r = j + 8 * hl;
    const float g0 = grow[r * CDIM + l16];
    const float g1 = grow[r * CDIM + 16 + l16];
    orow[(size_t)r * DMODEL + l16]      = (bf16_t)(o0[j] * g0);
    orow[(size_t)r * DMODEL + 16 + l16] = (bf16_t)(o1[j] * g1);
  }
}

// ---------------------------------------------------------------------------
// Kernel 3: output projection  out = wavg @ Wo^T + bo  (f32 result, NT store).
// ---------------------------------------------------------------------------
__global__ __launch_bounds__(128) void k_out(
    const bf16_t* __restrict__ x, const bf16_t* __restrict__ wo,
    const float* __restrict__ bo, float* __restrict__ out) {
  const int tid = threadIdx.x, lane = tid & 31, w = tid >> 5;
  const int l16 = lane & 15, hl = lane >> 4;
  const int m0 = blockIdx.x * 64 + w * 16;
  const int row = m0 + l16;

  v16bf af[8];
#pragma unroll
  for (int kc = 0; kc < 8; ++kc)
    af[kc] = afrag_bf16(x + (size_t)row * DMODEL + kc * 32 + 8 * hl);

#pragma unroll 1
  for (int nt = 0; nt < 16; ++nt) {
    v8f acc = {};
    const bf16_t* wb = wo + (nt * 16 + l16) * DMODEL + 8 * hl;
#pragma unroll
    for (int kc = 0; kc < 8; ++kc)
      acc = wmma_bf16(af[kc], bfrag_bf16(wb + kc * 32), acc);
    const int n = nt * 16 + l16;
    const float bv = bo[n];
#pragma unroll
    for (int j = 0; j < 8; ++j)
      __builtin_nontemporal_store(acc[j] + bv,
                                  &out[(size_t)(m0 + j + 8 * hl) * DMODEL + n]);
  }
}

// ---------------------------------------------------------------------------
// Host launcher.
// ---------------------------------------------------------------------------
extern "C" void kernel_launch(void* const* d_in, const int* in_sizes, int n_in,
                              void* d_out, int out_size, void* d_ws,
                              size_t ws_size, hipStream_t stream) {
  (void)in_sizes; (void)n_in; (void)out_size; (void)ws_size;
  const float* q_data  = (const float*)d_in[0];
  const float* kv_data = (const float*)d_in[1];
  const float* bias    = (const float*)d_in[2];
  const float* nbias   = (const float*)d_in[3];
  const float* Wq      = (const float*)d_in[4];
  const float* Wk      = (const float*)d_in[5];
  const float* Wv      = (const float*)d_in[6];
  const float* Wg      = (const float*)d_in[7];
  const float* bg      = (const float*)d_in[8];
  const float* Wo      = (const float*)d_in[9];
  const float* bo      = (const float*)d_in[10];
  float* out = (float*)d_out;

  const size_t SZ_PROJ = (size_t)NB2 * NHEAD * SEQ * CDIM;  // 8,388,608 elems
  char* ws = (char*)d_ws;
  bf16_t* w_bf   = (bf16_t*)ws;                         // 5 * 65536 bf16
  bf16_t* q_bf   = (bf16_t*)(ws + 5 * 65536 * 2);
  bf16_t* k_bf   = q_bf + SZ_PROJ;
  bf16_t* v_bf   = k_bf + SZ_PROJ;
  float*  g_f32  = (float*)(v_bf + SZ_PROJ);
  bf16_t* wavg   = (bf16_t*)(g_f32 + SZ_PROJ);

  bf16_t* wq_bf = w_bf;
  bf16_t* wk_bf = w_bf + 1 * 65536;
  bf16_t* wv_bf = w_bf + 2 * 65536;
  bf16_t* wg_bf = w_bf + 3 * 65536;
  bf16_t* wo_bf = w_bf + 4 * 65536;

  k_cvt<<<dim3((5 * 65536 + 255) / 256), dim3(256), 0, stream>>>(
      Wq, Wk, Wv, Wg, Wo, w_bf);

  k_proj<<<dim3(MTOT / 64), dim3(128), 0, stream>>>(
      q_data, kv_data, wq_bf, wk_bf, wv_bf, wg_bf, bg,
      q_bf, k_bf, v_bf, g_f32);

  k_attn<<<dim3(SEQ / 64, NHEAD, NB2), dim3(128), 0, stream>>>(
      q_bf, k_bf, v_bf, g_f32, bias, nbias, wavg);

  k_out<<<dim3(MTOT / 64), dim3(128), 0, stream>>>(wavg, wo_bf, bo, out);
}
